// LSTMClassifier_84713934946369
// MI455X (gfx1250) — matrix operand order repaired and checked by reading, not compiled, not run on hardware
//
#include <hip/hip_runtime.h>

// ---------------------------------------------------------------------------
// LSTM classifier for MI455X (gfx1250, wave32, WMMA).
//
// Roofline: 412 GFLOP total, x streams once (134 MB ~ 6us @ 23.3 TB/s),
// fp16 weights (3 MB) are L2-resident (192 MB L2). The recurrence is serial
// over T=512, so we minimize the per-step critical path: one fused GEMM
// C[256,2048] = [x_t | h][256,768] x Wcat^T via v_wmma_f32_16x16x32_f16,
// with the 4 gate tiles per wave so the cell update is done in-register.
//
// v2: the block's shared A tile (16 rows x 768 K, f16) is staged in LDS once
// per step (f32->f16 conversion done during the fill), removing the 8x
// duplicated global loads + converts across the block's waves. A fragments
// are then two ds_load_b128 per K-step.
// ---------------------------------------------------------------------------

typedef _Float16 half_t;
typedef __attribute__((ext_vector_type(16))) _Float16 v16h;
typedef __attribute__((ext_vector_type(8)))  _Float16 v8h;
typedef __attribute__((ext_vector_type(8)))  float    v8f;

#define INPUT  256
#define HIDDEN 512
#define G4     2048   // 4*HIDDEN
#define KTOT   768    // INPUT + HIDDEN
#define BSZ    256
#define TSZ    512
#define NCLS   128

// Workspace layout (bytes); total 4,333,568 B (~4.2 MB)
#define WCAT_OFF 0          // half[2048*768]  g-major: Wcat[g*768 + k]
#define WFC_OFF  3145728    // half[128*512]   Wfc[n*512 + k]
#define BIAS_OFF 3276800    // float[2048]     b_ih + b_hh
#define H16_OFF  3284992    // half[2][256*512] ping-pong h
#define C_OFF    3809280    // float[256*512]

// A-matrix (16x32 f16) K index of pair held in VGPR j for this lane half.
__device__ __forceinline__ int kpairA(int j, int half_) {
  return (j < 4) ? (half_ * 8 + 2 * j) : (16 + half_ * 8 + 2 * (j - 4));
}

__device__ __forceinline__ float sigmoidf_(float v) {
  return 1.0f / (1.0f + __expf(-v));
}
__device__ __forceinline__ float tanhf_(float v) {
  v = fminf(15.0f, fmaxf(-15.0f, v));
  float e = __expf(2.0f * v);
  return (e - 1.0f) / (e + 1.0f);
}

// ---------------------------------------------------------------------------
// One-time prep: fp32 -> fp16 weight repack + fused bias.
// Wcat[g][k] = (k<256 ? W_ih[g][k] : W_hh[g][k-256]); row-major per g so a
// B fragment (K block of 32, lane half picks K 8*half+[0,8) and 16+8*half+[0,8)) is two
// contiguous 16-byte loads.
// ---------------------------------------------------------------------------
__global__ void prep_kernel(const float* __restrict__ W_ih,
                            const float* __restrict__ W_hh,
                            const float* __restrict__ b_ih,
                            const float* __restrict__ b_hh,
                            const float* __restrict__ W_fc,
                            half_t* __restrict__ Wcat,
                            half_t* __restrict__ Wfc16,
                            float* __restrict__ bias) {
  const int NW = G4 * KTOT;
  const int NF = NCLS * HIDDEN;
  const int NB = G4;
  const int stride = gridDim.x * blockDim.x;
  for (int i = blockIdx.x * blockDim.x + threadIdx.x; i < NW + NF + NB; i += stride) {
    if (i < NW) {
      int g = i / KTOT, k = i - g * KTOT;
      float w = (k < INPUT) ? W_ih[g * INPUT + k] : W_hh[g * HIDDEN + (k - INPUT)];
      Wcat[i] = (half_t)w;
    } else if (i < NW + NF) {
      int j = i - NW;
      Wfc16[j] = (half_t)W_fc[j];
    } else {
      int g = i - NW - NF;
      bias[g] = b_ih[g] + b_hh[g];
    }
  }
}

__global__ void init_kernel(half_t* __restrict__ h16, float* __restrict__ c) {
  int i = blockIdx.x * blockDim.x + threadIdx.x;
  if (i < BSZ * HIDDEN) { h16[i] = (half_t)0.0f; c[i] = 0.0f; }
}

// ---------------------------------------------------------------------------
// One LSTM timestep. 64 blocks x 8 waves = 512 waves. All waves of a block
// share the same 16-row A tile (staged in LDS); wave w owns gate-column tile
// nt = (blockIdx&3)*8 + w, computed for all four gates i,f,g,o, then the
// cell update runs in-register on the accumulator fragments.
// ---------------------------------------------------------------------------
__global__ void __launch_bounds__(256)
lstm_step_kernel(const float* __restrict__ x, int t,
                 const half_t* __restrict__ Wcat,
                 const float* __restrict__ bias,
                 const half_t* __restrict__ h_in,
                 half_t* __restrict__ h_out,
                 float* __restrict__ c) {
  __shared__ __align__(16) half_t As[16 * KTOT];   // 24 KB of the 320KB WGP LDS

  const int tid   = threadIdx.x;
  const int lane  = tid & 31;
  const int wave  = tid >> 5;                        // 0..7
  const int mt    = blockIdx.x >> 2;                 // 0..15 (shared by block)
  const int nt    = ((blockIdx.x & 3) << 3) + wave;  // 0..31
  const int n     = lane & 15;
  const int half_ = lane >> 4;
  const int m_base = mt * 16;
  const int n_base = nt * 16;

  // ---- Cooperative LDS fill: As[row][0:256) = f16(x[m_base+row, t, :]),
  //      As[row][256:768) = h_in[m_base+row, :]  (already f16) ----
  // x part: 16x256 f32 -> f16, 8 elements per iteration (2 x float4)
  for (int i = tid; i < (16 * INPUT) / 8; i += 256) {
    const int e   = i * 8;
    const int row = e >> 8;          // /256
    const int k   = e & 255;
    const float* src = x + (size_t)(m_base + row) * (TSZ * INPUT)
                         + (size_t)t * INPUT + k;
    const float4 x0 = *(const float4*)(src);
    const float4 x1 = *(const float4*)(src + 4);
    v8h v;
    v[0] = (half_t)x0.x; v[1] = (half_t)x0.y;
    v[2] = (half_t)x0.z; v[3] = (half_t)x0.w;
    v[4] = (half_t)x1.x; v[5] = (half_t)x1.y;
    v[6] = (half_t)x1.z; v[7] = (half_t)x1.w;
    *(v8h*)(As + row * KTOT + k) = v;
  }
  // h part: 16x512 f16 copy, 8 halfs (16B) per iteration
  for (int i = tid; i < (16 * HIDDEN) / 8; i += 256) {
    const int e   = i * 8;
    const int row = e >> 9;          // /512
    const int k   = e & 511;
    *(v8h*)(As + row * KTOT + INPUT + k) =
        *(const v8h*)(h_in + (size_t)(m_base + row) * HIDDEN + k);
  }
  __syncthreads();

  const half_t* wrow[4];
  v8f acc[4];
#pragma unroll
  for (int gi = 0; gi < 4; ++gi) {
    const int gcol = gi * HIDDEN + n_base + n;
    wrow[gi] = Wcat + (size_t)gcol * KTOT;
    const float bv = bias[gcol];
#pragma unroll
    for (int j = 0; j < 8; ++j) acc[gi][j] = bv;
  }

  // A row held by this lane (A-matrix row = lane % 16)
  const half_t* arow = As + n * KTOT;

  // ---- Single fused K loop over [x_t | h] (24 steps of K=32) ----
  for (int kk = 0; kk < KTOT / 32; ++kk) {
    const int k0 = kk * 32;
    // A fragment = two contiguous 16B runs in LDS (matches ISA A layout:
    // VGPRs 0-3 hold K = 8*half + [0,8), VGPRs 4-7 hold K = 16 + 8*half + [0,8))
    const v8h lo = *(const v8h*)(arow + k0 + half_ * 8);
    const v8h hi = *(const v8h*)(arow + k0 + 16 + half_ * 8);
    v16h a;
#pragma unroll
    for (int j = 0; j < 8; ++j) { a[j] = lo[j]; a[8 + j] = hi[j]; }

#pragma unroll
    for (int gi = 0; gi < 4; ++gi) {
      // B 32x16: same K order as A -> two 16B loads (K = 8*half + [0,8), 16 + 8*half + [0,8))
      const v8h blo = *(const v8h*)(wrow[gi] + k0 + half_ * 8);
      const v8h bhi = *(const v8h*)(wrow[gi] + k0 + 16 + half_ * 8);
      v16h b;
#pragma unroll
      for (int j = 0; j < 8; ++j) { b[j] = blo[j]; b[8 + j] = bhi[j]; }
      acc[gi] = __builtin_amdgcn_wmma_f32_16x16x32_f16(
          false, a, false, b, (short)0, acc[gi], false, false);
    }
  }

  // ---- In-register LSTM cell update (lane owns rows half*8+j, col n) ----
  const int col = n_base + n;
#pragma unroll
  for (int j = 0; j < 8; ++j) {
    const int row = m_base + half_ * 8 + j;
    const int idx = row * HIDDEN + col;
    const float iv = sigmoidf_(acc[0][j]);
    const float fv = sigmoidf_(acc[1][j]);
    const float gv = tanhf_(acc[2][j]);
    const float ov = sigmoidf_(acc[3][j]);
    const float cv = fv * c[idx] + iv * gv;
    c[idx]     = cv;
    h_out[idx] = (half_t)(ov * tanhf_(cv));
  }
}

// ---------------------------------------------------------------------------
// Final FC: out[256,128] = h_last @ W_fc^T + b_fc. 128 waves, one tile each.
// ---------------------------------------------------------------------------
__global__ void __launch_bounds__(256)
fc_kernel(const half_t* __restrict__ h16, const half_t* __restrict__ Wfc,
          const float* __restrict__ b_fc, float* __restrict__ out) {
  const int lane  = threadIdx.x & 31;
  const int wid   = (blockIdx.x * blockDim.x + threadIdx.x) >> 5;  // 0..127
  const int mt    = wid >> 3;   // 0..15
  const int nt    = wid & 7;    // 0..7
  const int n     = lane & 15;
  const int half_ = lane >> 4;
  const int m_base = mt * 16;
  const int n_base = nt * 16;

  const half_t* hrow = h16 + (size_t)(m_base + n) * HIDDEN;
  const half_t* wr   = Wfc + (size_t)(n_base + n) * HIDDEN;

  v8f acc;
  const float bv = b_fc[n_base + n];
#pragma unroll
  for (int j = 0; j < 8; ++j) acc[j] = bv;

  for (int kk = 0; kk < HIDDEN / 32; ++kk) {
    const int k0 = kk * 32;
    const v8h lo = *(const v8h*)(hrow + k0 + half_ * 8);
    const v8h hi = *(const v8h*)(hrow + k0 + 16 + half_ * 8);
    v16h a;
#pragma unroll
    for (int j = 0; j < 8; ++j) { a[j] = lo[j]; a[8 + j] = hi[j]; }
    const v8h blo = *(const v8h*)(wr + k0 + half_ * 8);
    const v8h bhi = *(const v8h*)(wr + k0 + 16 + half_ * 8);
    v16h b;
#pragma unroll
    for (int j = 0; j < 8; ++j) { b[j] = blo[j]; b[8 + j] = bhi[j]; }
    acc = __builtin_amdgcn_wmma_f32_16x16x32_f16(
        false, a, false, b, (short)0, acc, false, false);
  }

  const int col = n_base + n;
#pragma unroll
  for (int j = 0; j < 8; ++j) {
    const int row = m_base + half_ * 8 + j;
    out[row * NCLS + col] = acc[j];
  }
}

// ---------------------------------------------------------------------------
extern "C" void kernel_launch(void* const* d_in, const int* in_sizes, int n_in,
                              void* d_out, int out_size, void* d_ws, size_t ws_size,
                              hipStream_t stream) {
  (void)in_sizes; (void)n_in; (void)out_size; (void)ws_size;
  const float* x    = (const float*)d_in[0];
  const float* W_ih = (const float*)d_in[1];
  const float* W_hh = (const float*)d_in[2];
  const float* b_ih = (const float*)d_in[3];
  const float* b_hh = (const float*)d_in[4];
  const float* W_fc = (const float*)d_in[5];
  const float* b_fc = (const float*)d_in[6];

  char* ws = (char*)d_ws;   // requires ws_size >= 4,333,568 bytes (~4.2 MB)
  half_t* Wcat  = (half_t*)(ws + WCAT_OFF);
  half_t* Wfc16 = (half_t*)(ws + WFC_OFF);
  float*  bias  = (float*) (ws + BIAS_OFF);
  half_t* h16   = (half_t*)(ws + H16_OFF);   // two ping-pong buffers
  float*  c     = (float*) (ws + C_OFF);

  prep_kernel<<<2048, 256, 0, stream>>>(W_ih, W_hh, b_ih, b_hh, W_fc,
                                        Wcat, Wfc16, bias);
  init_kernel<<<(BSZ * HIDDEN + 255) / 256, 256, 0, stream>>>(h16, c);

  for (int t = 0; t < TSZ; ++t) {
    half_t* hin  = h16 + (size_t)(t & 1) * (BSZ * HIDDEN);
    half_t* hout = h16 + (size_t)((t + 1) & 1) * (BSZ * HIDDEN);
    lstm_step_kernel<<<64, 256, 0, stream>>>(x, t, Wcat, bias, hin, hout, c);
  }
  // T=512 is even -> h_last lands in buffer 0
  fc_kernel<<<16, 256, 0, stream>>>(h16, Wfc16, b_fc, (float*)d_out);
}
